// LSTM_Global_Attention_32238024524120
// MI455X (gfx1250) — matrix operand order repaired and checked by reading, not compiled, not run on hardware
//
#include <hip/hip_runtime.h>
#include <hip/hip_bf16.h>
#include <cstdint>
#include <cstddef>

typedef __bf16 bf16_t;
typedef __attribute__((ext_vector_type(16))) __bf16 v16bf;
typedef __attribute__((ext_vector_type(8)))  __bf16 v8bf;
typedef __attribute__((ext_vector_type(8)))  float  v8f;
typedef __attribute__((ext_vector_type(2)))  int    v2i;

#define DEVI __device__ __forceinline__

// ---- gfx1250 async memory->LDS copy (guarded; falls back to VGPR staging)
#if defined(__has_builtin)
#  if __has_builtin(__builtin_amdgcn_global_load_async_to_lds_b64)
#    define HAVE_ASYNC_COPY 1
#  endif
#  if __has_builtin(__builtin_amdgcn_s_wait_asynccnt)
#    define HAVE_WAIT_ASYNC 1
#  endif
#endif

// builtin signature (from hipcc diagnostic): (v2i32 AS1*, v2i32 AS3*, Ii, Ii)
typedef __attribute__((address_space(1))) v2i* as1_v2i;
typedef __attribute__((address_space(3))) v2i* as3_v2i;

DEVI void stage8(const bf16_t* g, bf16_t* l) {
#ifdef HAVE_ASYNC_COPY
  __builtin_amdgcn_global_load_async_to_lds_b64((as1_v2i)(void*)g,
                                                (as3_v2i)(void*)l, 0, 0);
#else
  *(uint2*)l = *(const uint2*)g;
#endif
}
DEVI void async_fence() {
#ifdef HAVE_ASYNC_COPY
#  ifdef HAVE_WAIT_ASYNC
  __builtin_amdgcn_s_wait_asynccnt(0);
#  else
  asm volatile("s_wait_asynccnt 0x0" ::: "memory");
#  endif
#endif
}

DEVI v8f v8f_zero() {
  v8f v;
#pragma unroll
  for (int i = 0; i < 8; ++i) v[i] = 0.0f;
  return v;
}

DEVI v16bf frag_from(const bf16_t* p0, const bf16_t* p1) {
  v8bf lo = *(const v8bf*)p0;
  v8bf hi = *(const v8bf*)p1;
  v16bf f;
#pragma unroll
  for (int i = 0; i < 8; ++i) { f[i] = lo[i]; f[i + 8] = hi[i]; }
  return f;
}

// A fragment (16x32 tile of row-major [M,K], per CDNA5 16-bit A layout):
// lane L: row = L&15; K-halves at (L>>4)*8 and (L>>4)*8+16
DEVI v16bf fragA(const bf16_t* tile, int stride, int lane) {
  const bf16_t* p = tile + (size_t)(lane & 15) * stride + ((lane >> 4) * 8);
  return frag_from(p, p + 16);
}
// B fragment: weights stored [N,K] row-major (computing A @ W^T):
// lane L: col N = L&15; K-halves at (L>>4)*8 and (L>>4)*8+16
DEVI v16bf fragB(const bf16_t* w, int ldw, int lane) {
  const bf16_t* p = w + (size_t)(lane & 15) * ldw + ((lane >> 4) * 8);
  return frag_from(p, p + 16);
}

DEVI v8f wmma_bf16(v16bf a, v16bf b, v8f c) {
  return __builtin_amdgcn_wmma_f32_16x16x32_bf16(false, a, false, b, (short)0, c,
                                                 false, false);
}

DEVI float sigm(float x) { return 1.0f / (1.0f + __expf(-x)); }

// ---------------------------------------------------------------- utilities
__global__ void k_cast(const float* __restrict__ s, bf16_t* __restrict__ d, int n) {
  for (int i = blockIdx.x * 256 + threadIdx.x; i < n; i += gridDim.x * 256)
    d[i] = (bf16_t)s[i];
}

__global__ void k_zero_bf16(bf16_t* p, int n) {
  int i = blockIdx.x * 256 + threadIdx.x;
  if (i < n) p[i] = (bf16_t)0.0f;
}

__global__ void k_src_lens(const int* __restrict__ x, int* __restrict__ lens) {
  int b = threadIdx.x;
  if (b < 32) {
    int n = 0;
    for (int s = 0; s < 64; ++s) n += (x[b * 64 + s] != 0);
    lens[b] = n;
  }
}

// source: reverse valid prefix + gather embedding -> bf16 [S*B, 512]
__global__ void k_src_embed(const int* __restrict__ x, const float* __restrict__ emb,
                            const int* __restrict__ lens, bf16_t* __restrict__ out) {
  int blk = blockIdx.x;            // s*32 + b
  int s = blk >> 5, b = blk & 31;
  int len = lens[b];
  int idx = (s < len) ? (len - 1 - s) : s;
  int tok = x[b * 64 + idx];
  const float* e = emb + (size_t)tok * 512;
  bf16_t* o = out + (size_t)blk * 512;
  for (int i = threadIdx.x; i < 512; i += 256) o[i] = (bf16_t)e[i];
}

// target: gather embedding for y[:, :-1] -> bf16 [(T-1)*B, 512]
__global__ void k_tgt_embed(const int* __restrict__ y, const float* __restrict__ emb,
                            bf16_t* __restrict__ out) {
  int blk = blockIdx.x;            // t*32 + b, t in [0,63)
  int t = blk >> 5, b = blk & 31;
  int tok = y[b * 64 + t];
  const float* e = emb + (size_t)tok * 512;
  bf16_t* o = out + (size_t)blk * 512;
  for (int i = threadIdx.x; i < 512; i += 256) o[i] = (bf16_t)e[i];
}

// ------------------------------------------------ generic big GEMM (WMMA)
// C[M,N](f32) = A[M,K](bf16,row-major) @ W[N,K](bf16,row-major)^T (+bias)
// block 256 thr (8 waves), BLOCK_M=32 BLOCK_N=128; grid=(N/128, M/32); K%32==0
// A tile double-buffered in LDS via async copy: one barrier per K-step.
__global__ void __launch_bounds__(256)
k_gemm_xw(const bf16_t* __restrict__ A, int lda,
          const bf16_t* __restrict__ W, int ldw,
          const float* __restrict__ bias,
          float* __restrict__ C, int ldc, int K) {
  __shared__ bf16_t As[2][32 * 32];
  const int tid = threadIdx.x, lane = tid & 31, wid = tid >> 5;
  const int m0 = blockIdx.y * 32;
  const int n0 = blockIdx.x * 128 + wid * 16;
  const int r = (tid * 4) >> 5, c = (tid * 4) & 31;
  const bf16_t* aSrc = A + (size_t)(m0 + r) * lda + c;
  const bf16_t* wRow = W + (size_t)(n0 + (lane & 15)) * ldw;

  stage8(aSrc, &As[0][r * 32 + c]);
  async_fence();
  __syncthreads();

  v8f acc0 = v8f_zero(), acc1 = v8f_zero();
  for (int k0 = 0; k0 < K; k0 += 32) {
    int cur = (k0 >> 5) & 1;
    if (k0 + 32 < K) {
      stage8(aSrc + k0 + 32, &As[cur ^ 1][r * 32 + c]);
      __builtin_prefetch(wRow + k0 + 32, 0, 1);   // next W K-block
    }
    v16bf b  = fragB(W + (size_t)n0 * ldw + k0, ldw, lane);
    v16bf a0 = fragA(As[cur], 32, lane);
    v16bf a1 = fragA(As[cur] + 16 * 32, 32, lane);
    acc0 = wmma_bf16(a0, b, acc0);
    acc1 = wmma_bf16(a1, b, acc1);
    async_fence();
    __syncthreads();
  }
  int col = n0 + (lane & 15);
  float bv = bias ? bias[col] : 0.0f;
#pragma unroll
  for (int i = 0; i < 8; ++i) {
    int r0 = m0 + ((lane >> 4) * 8) + i;
    C[(size_t)r0 * ldc + col]        = acc0[i] + bv;
    C[(size_t)(r0 + 16) * ldc + col] = acc1[i] + bv;
  }
}

// --------------------------------------- persistent bidirectional LSTM scan
// grid=2 (dir), block=512 (16 waves). H=512, gates 4x512, K=512, B=32.
// Xg*: precomputed x@Wih^T f32 [64*32, 2048]. h double-buffered in LDS,
// c lives in accumulator registers across all 64 steps. One barrier/step.
__global__ void __launch_bounds__(512)
k_enc_scan(const float* __restrict__ Xg0, const float* __restrict__ Xg1,
           const bf16_t* __restrict__ Whh0, const bf16_t* __restrict__ Whh1,
           const float* __restrict__ bih0, const float* __restrict__ bhh0,
           const float* __restrict__ bih1, const float* __restrict__ bhh1,
           bf16_t* __restrict__ O,      // [64][32][1024], this dir -> cols dir*512..
           bf16_t* __restrict__ decH,   // [32][1024]
           float*  __restrict__ decC) { // [32][1024]
  __shared__ bf16_t hbuf[2][32][512];
  const int dir = blockIdx.x;
  const float*  Xg  = dir ? Xg1  : Xg0;
  const bf16_t* Whh = dir ? Whh1 : Whh0;
  const float*  bih = dir ? bih1 : bih0;
  const float*  bhh = dir ? bhh1 : bhh0;
  const int tid = threadIdx.x, lane = tid & 31, wid = tid >> 5;

  bf16_t* hz = &hbuf[0][0][0];
  for (int i = tid; i < 32 * 512; i += 512) hz[i] = (bf16_t)0.0f;
  __syncthreads();

  int mt[4], nt[4];
  v8f cst[4];
  float bsum[4][4];
#pragma unroll
  for (int j = 0; j < 4; ++j) {
    int u = wid * 4 + j;
    mt[j] = u & 1; nt[j] = u >> 1;
    cst[j] = v8f_zero();
    int col = nt[j] * 16 + (lane & 15);
#pragma unroll
    for (int g = 0; g < 4; ++g)
      bsum[j][g] = bih[g * 512 + col] + bhh[g * 512 + col];
  }

  for (int step = 0; step < 64; ++step) {
    int t = dir ? (63 - step) : step;
    int cur = step & 1, nxt = cur ^ 1;
#pragma unroll
    for (int j = 0; j < 4; ++j) {
      v8f acc[4];
#pragma unroll
      for (int g = 0; g < 4; ++g) acc[g] = v8f_zero();
      for (int ks = 0; ks < 512; ks += 32) {
        v16bf a = fragA(&hbuf[cur][mt[j] * 16][ks], 512, lane);
#pragma unroll
        for (int g = 0; g < 4; ++g) {
          v16bf b = fragB(Whh + (size_t)(g * 512 + nt[j] * 16) * 512 + ks, 512, lane);
          acc[g] = wmma_bf16(a, b, acc[g]);
        }
      }
      int col = nt[j] * 16 + (lane & 15);
#pragma unroll
      for (int i = 0; i < 8; ++i) {
        int row = mt[j] * 16 + ((lane >> 4) * 8) + i;
        size_t xb = ((size_t)t * 32 + row) * 2048;
        float gi = sigm (acc[0][i] + Xg[xb +    0 + col] + bsum[j][0]);
        float gf = sigm (acc[1][i] + Xg[xb +  512 + col] + bsum[j][1]);
        float gg = tanhf(acc[2][i] + Xg[xb + 1024 + col] + bsum[j][2]);
        float go = sigm (acc[3][i] + Xg[xb + 1536 + col] + bsum[j][3]);
        float cn = gf * cst[j][i] + gi * gg;
        cst[j][i] = cn;
        float hn = go * tanhf(cn);
        bf16_t hb = (bf16_t)hn;
        hbuf[nxt][row][col] = hb;
        O[((size_t)t * 32 + row) * 1024 + dir * 512 + col] = hb;
        if (step == 63) {
          decH[(size_t)row * 1024 + dir * 512 + col] = hb;
          decC[(size_t)row * 1024 + dir * 512 + col] = cn;
        }
      }
    }
    __syncthreads();
  }
}

// -------------------------------------------------- decoder LSTM cell step
struct GemmSrc { const bf16_t* A; int lda; const bf16_t* W; int ldw; int K; };

// gates[32,4096] = sum_d A_d @ W_d^T ; then LSTM elementwise.
// grid=16, block=256 (8 waves): 128 wave-units = mt(2) x nt(64); each unit
// owns i/f/g/o tiles at the same (row,col) so no gate scratch is needed.
__global__ void __launch_bounds__(256)
k_dec_cell(GemmSrc s0, GemmSrc s1, GemmSrc s2,
           const float* __restrict__ bih, const float* __restrict__ bhh,
           float* __restrict__ cstate, bf16_t* __restrict__ hout) {
  const int tid = threadIdx.x, lane = tid & 31, wid = tid >> 5;
  const int u = blockIdx.x * 8 + wid;
  const int mt = u & 1, nt = u >> 1;
  v8f acc[4];
#pragma unroll
  for (int g = 0; g < 4; ++g) acc[g] = v8f_zero();
  GemmSrc ss[3] = {s0, s1, s2};
#pragma unroll
  for (int d = 0; d < 3; ++d) {
    if (!ss[d].A) continue;
    for (int k0 = 0; k0 < ss[d].K; k0 += 32) {
      v16bf a = fragA(ss[d].A + (size_t)(mt * 16) * ss[d].lda + k0, ss[d].lda, lane);
#pragma unroll
      for (int g = 0; g < 4; ++g) {
        v16bf b = fragB(ss[d].W + (size_t)(g * 1024 + nt * 16) * ss[d].ldw + k0,
                        ss[d].ldw, lane);
        acc[g] = wmma_bf16(a, b, acc[g]);
      }
    }
  }
  int col = nt * 16 + (lane & 15);
  float bs[4];
#pragma unroll
  for (int g = 0; g < 4; ++g) bs[g] = bih[g * 1024 + col] + bhh[g * 1024 + col];
#pragma unroll
  for (int i = 0; i < 8; ++i) {
    int row = mt * 16 + ((lane >> 4) * 8) + i;
    float gi = sigm (acc[0][i] + bs[0]);
    float gf = sigm (acc[1][i] + bs[1]);
    float gg = tanhf(acc[2][i] + bs[2]);
    float go = sigm (acc[3][i] + bs[3]);
    float cn = gf * cstate[row * 1024 + col] + gi * gg;
    cstate[row * 1024 + col] = cn;
    hout[row * 1024 + col] = (bf16_t)(go * tanhf(cn));
  }
}

// ---------------------------------------- small M=32 GEMM (WMMA), 2 sources
__global__ void __launch_bounds__(256)
k_gemm32(GemmSrc s0, GemmSrc s1,
         float* __restrict__ Cf, int ldc,
         bf16_t* __restrict__ Cb0, int ldb0,
         bf16_t* __restrict__ Cb1, long ldb1, int do_tanh) {
  const int tid = threadIdx.x, lane = tid & 31, wid = tid >> 5;
  const int n0 = (blockIdx.x * 8 + wid) * 16;
  v8f acc0 = v8f_zero(), acc1 = v8f_zero();
  GemmSrc ss[2] = {s0, s1};
#pragma unroll
  for (int d = 0; d < 2; ++d) {
    if (!ss[d].A) continue;
    for (int k0 = 0; k0 < ss[d].K; k0 += 32) {
      v16bf b  = fragB(ss[d].W + (size_t)n0 * ss[d].ldw + k0, ss[d].ldw, lane);
      v16bf a0 = fragA(ss[d].A + k0, ss[d].lda, lane);
      v16bf a1 = fragA(ss[d].A + (size_t)16 * ss[d].lda + k0, ss[d].lda, lane);
      acc0 = wmma_bf16(a0, b, acc0);
      acc1 = wmma_bf16(a1, b, acc1);
    }
  }
  int col = n0 + (lane & 15);
#pragma unroll
  for (int i = 0; i < 8; ++i) {
    int r0 = ((lane >> 4) * 8) + i, r1 = r0 + 16;
    float v0 = acc0[i], v1 = acc1[i];
    if (do_tanh) { v0 = tanhf(v0); v1 = tanhf(v1); }
    if (Cf)  { Cf[r0 * ldc + col] = v0;  Cf[r1 * ldc + col] = v1; }
    if (Cb0) { Cb0[r0 * ldb0 + col] = (bf16_t)v0; Cb0[r1 * ldb0 + col] = (bf16_t)v1; }
    if (Cb1) { Cb1[(size_t)r0 * ldb1 + col] = (bf16_t)v0;
               Cb1[(size_t)r1 * ldb1 + col] = (bf16_t)v1; }
  }
}

// ------------------------------------------- attention: scores/softmax/ctx
__global__ void __launch_bounds__(256)
k_attn(const float* __restrict__ hWa, const bf16_t* __restrict__ enc,
       const int* __restrict__ lens, bf16_t* __restrict__ ctx) {
  __shared__ float part[64][4];
  __shared__ float sc[64];
  __shared__ float red;
  const int b = blockIdx.x, tid = threadIdx.x;
  int s = tid >> 2, q = tid & 3;
  const float*  hv = hWa + b * 1024;
  const bf16_t* ev = enc + ((size_t)s * 32 + b) * 1024;
  float p = 0.f;
  for (int d = q * 256; d < q * 256 + 256; ++d) p += hv[d] * (float)ev[d];
  part[s][q] = p;
  __syncthreads();
  if (tid < 64) {
    float v = part[tid][0] + part[tid][1] + part[tid][2] + part[tid][3];
    sc[tid] = (tid < lens[b]) ? v : -1e9f;
  }
  __syncthreads();
  if (tid == 0) {
    float mx = sc[0];
    for (int i = 1; i < 64; ++i) mx = fmaxf(mx, sc[i]);
    float sum = 0.f;
    for (int i = 0; i < 64; ++i) { sc[i] = __expf(sc[i] - mx); sum += sc[i]; }
    red = 1.0f / sum;
  }
  __syncthreads();
  float inv = red;
  for (int d = tid; d < 1024; d += 256) {
    float a = 0.f;
    for (int s2 = 0; s2 < 64; ++s2)
      a += sc[s2] * (float)enc[((size_t)s2 * 32 + b) * 1024 + d];
    ctx[b * 1024 + d] = (bf16_t)(a * inv);
  }
}

// ======================================================================
extern "C" void kernel_launch(void* const* d_in, const int* in_sizes, int n_in,
                              void* d_out, int out_size, void* d_ws, size_t ws_size,
                              hipStream_t stream) {
  (void)in_sizes; (void)n_in; (void)out_size; (void)ws_size;
  const int* x = (const int*)d_in[0];
  const int* y = (const int*)d_in[1];
  const float* src_emb = (const float*)d_in[2];
  const float* tgt_emb = (const float*)d_in[3];
  // enc: l0f(4..7) l0b(8..11) l1f(12..15) l1b(16..19); dec0(20..23) dec1(24..27)
  const float* bih_l0f = (const float*)d_in[6],  *bhh_l0f = (const float*)d_in[7];
  const float* bih_l0b = (const float*)d_in[10], *bhh_l0b = (const float*)d_in[11];
  const float* bih_l1f = (const float*)d_in[14], *bhh_l1f = (const float*)d_in[15];
  const float* bih_l1b = (const float*)d_in[18], *bhh_l1b = (const float*)d_in[19];
  const float* bih_d0  = (const float*)d_in[22], *bhh_d0  = (const float*)d_in[23];
  const float* bih_d1  = (const float*)d_in[26], *bhh_d1  = (const float*)d_in[27];
  const float* fc_b    = (const float*)d_in[31];

  char* ws = (char*)d_ws;
  size_t off = 0;
  auto alloc = [&](size_t bytes) -> void* {
    void* p = ws + off;
    off = (off + bytes + 255) & ~(size_t)255;
    return p;
  };
  auto cast = [&](const void* src, size_t n) -> bf16_t* {
    bf16_t* d = (bf16_t*)alloc(n * sizeof(bf16_t));
    k_cast<<<256, 256, 0, stream>>>((const float*)src, d, (int)n);
    return d;
  };

  // bf16 weights (stored [N,K] just like the inputs; B-fragment-friendly)
  bf16_t* wih_l0f = cast(d_in[4],  2048 * 512);
  bf16_t* whh_l0f = cast(d_in[5],  2048 * 512);
  bf16_t* wih_l0b = cast(d_in[8],  2048 * 512);
  bf16_t* whh_l0b = cast(d_in[9],  2048 * 512);
  bf16_t* wih_l1f = cast(d_in[12], 2048 * 1024);
  bf16_t* whh_l1f = cast(d_in[13], 2048 * 512);
  bf16_t* wih_l1b = cast(d_in[16], 2048 * 1024);
  bf16_t* whh_l1b = cast(d_in[17], 2048 * 512);
  bf16_t* wih_d0  = cast(d_in[20], 4096 * 1536);
  bf16_t* whh_d0  = cast(d_in[21], 4096 * 1024);
  bf16_t* wih_d1  = cast(d_in[24], 4096 * 1024);
  bf16_t* whh_d1  = cast(d_in[25], 4096 * 1024);
  bf16_t* wa_bf   = cast(d_in[28], 1024 * 1024);
  bf16_t* wc_bf   = cast(d_in[29], 1024 * 2048);
  bf16_t* fcw_bf  = cast(d_in[30], (size_t)32000 * 1024);

  bf16_t* Xemb  = (bf16_t*)alloc((size_t)2048 * 512 * 2);
  bf16_t* Yemb  = (bf16_t*)alloc((size_t)63 * 32 * 512 * 2);
  bf16_t* O0    = (bf16_t*)alloc((size_t)64 * 32 * 1024 * 2);
  bf16_t* encO  = (bf16_t*)alloc((size_t)64 * 32 * 1024 * 2);
  float*  XgF   = (float*)alloc((size_t)2048 * 2048 * 4);
  float*  XgB   = (float*)alloc((size_t)2048 * 2048 * 4);
  bf16_t* HtAll = (bf16_t*)alloc((size_t)2016 * 1024 * 2);
  bf16_t* h0b[2] = {(bf16_t*)alloc(32 * 1024 * 2), (bf16_t*)alloc(32 * 1024 * 2)};
  bf16_t* h1b[2] = {(bf16_t*)alloc(32 * 1024 * 2), (bf16_t*)alloc(32 * 1024 * 2)};
  bf16_t* ht    = (bf16_t*)alloc(32 * 1024 * 2);
  bf16_t* ctx   = (bf16_t*)alloc(32 * 1024 * 2);
  float*  c0    = (float*)alloc(32 * 1024 * 4);
  float*  c1    = (float*)alloc(32 * 1024 * 4);
  float*  hWa   = (float*)alloc(32 * 1024 * 4);
  int*    lens  = (int*)alloc(32 * 4);

  // ---- embeddings
  k_src_lens<<<1, 64, 0, stream>>>(x, lens);
  k_src_embed<<<2048, 256, 0, stream>>>(x, src_emb, lens, Xemb);
  k_tgt_embed<<<63 * 32, 256, 0, stream>>>(y, tgt_emb, Yemb);

  // ---- encoder layer 0
  k_gemm_xw<<<dim3(16, 64), 256, 0, stream>>>(Xemb, 512, wih_l0f, 512, nullptr,
                                              XgF, 2048, 512);
  k_gemm_xw<<<dim3(16, 64), 256, 0, stream>>>(Xemb, 512, wih_l0b, 512, nullptr,
                                              XgB, 2048, 512);
  k_enc_scan<<<2, 512, 0, stream>>>(XgF, XgB, whh_l0f, whh_l0b,
                                    bih_l0f, bhh_l0f, bih_l0b, bhh_l0b,
                                    O0, h0b[0], c0);
  // ---- encoder layer 1
  k_gemm_xw<<<dim3(16, 64), 256, 0, stream>>>(O0, 1024, wih_l1f, 1024, nullptr,
                                              XgF, 2048, 1024);
  k_gemm_xw<<<dim3(16, 64), 256, 0, stream>>>(O0, 1024, wih_l1b, 1024, nullptr,
                                              XgB, 2048, 1024);
  k_enc_scan<<<2, 512, 0, stream>>>(XgF, XgB, whh_l1f, whh_l1b,
                                    bih_l1f, bhh_l1f, bih_l1b, bhh_l1b,
                                    encO, h1b[0], c1);

  // ---- decoder
  k_zero_bf16<<<(32 * 1024 + 255) / 256, 256, 0, stream>>>(ht, 32 * 1024);
  GemmSrc zs = {nullptr, 0, nullptr, 0, 0};
  for (int t = 0; t < 63; ++t) {
    int i0 = t & 1, o0 = i0 ^ 1;
    GemmSrc se = {Yemb + (size_t)t * 32 * 512, 512, wih_d0, 1536, 512};
    GemmSrc sh = {ht, 1024, wih_d0 + 512, 1536, 1024};
    GemmSrc sr = {h0b[i0], 1024, whh_d0, 1024, 1024};
    k_dec_cell<<<16, 256, 0, stream>>>(se, sh, sr, bih_d0, bhh_d0, c0, h0b[o0]);

    GemmSrc s10 = {h0b[o0], 1024, wih_d1, 1024, 1024};
    GemmSrc s11 = {h1b[i0], 1024, whh_d1, 1024, 1024};
    k_dec_cell<<<16, 256, 0, stream>>>(s10, s11, zs, bih_d1, bhh_d1, c1, h1b[o0]);

    GemmSrc sa = {h1b[o0], 1024, wa_bf, 1024, 1024};
    k_gemm32<<<8, 256, 0, stream>>>(sa, zs, hWa, 1024, nullptr, 0, nullptr, 0, 0);

    k_attn<<<32, 256, 0, stream>>>(hWa, encO, lens, ctx);

    GemmSrc sc0 = {ctx, 1024, wc_bf, 2048, 1024};
    GemmSrc sc1 = {h1b[o0], 1024, wc_bf + 1024, 2048, 1024};
    k_gemm32<<<8, 256, 0, stream>>>(sc0, sc1, nullptr, 0, ht, 1024,
                                    HtAll + (size_t)t * 1024, (long)63 * 1024, 1);
  }

  // ---- output projection: [2016,1024] @ [32000,1024]^T + fc_b -> d_out
  k_gemm_xw<<<dim3(250, 63), 256, 0, stream>>>(HtAll, 1024, fcw_bf, 1024, fc_b,
                                               (float*)d_out, 32000, 1024);
}
